// EdgeAsAttendeeSelfAttention_6322191860304
// MI455X (gfx1250) — matrix operand order repaired and checked by reading, not compiled, not run on hardware
//
#include <hip/hip_runtime.h>
#include <hip/hip_bf16.h>

// ---------------------------------------------------------------------------
// Problem constants (match reference):
//   B=8, N=256, DEG=32, H=768, NH=12, dh=64, R=64, E = B*N*DEG = 65536
// ---------------------------------------------------------------------------
#define BQ   8
#define NN   256
#define DEG  32
#define HH   768
#define NHD  12
#define DH   64
#define EE   (BQ * NN * DEG)
#define MROWS (BQ * NN)            // 2048 node rows
#define WELEM (HH * HH)            // 589824 weight elements

typedef __bf16 v16bf __attribute__((ext_vector_type(16)));
typedef __bf16 v8bf  __attribute__((ext_vector_type(8)));
typedef __bf16 v4bf  __attribute__((ext_vector_type(4)));
typedef float  v8f   __attribute__((ext_vector_type(8)));

// ---------------------------------------------------------------------------
// CDNA5 async global->LDS copy (ISA 15.18.3, ASYNCcnt-tracked).
// GVS mode: uniform SGPR base + per-lane VGPR byte offset. VDST is the LDS
// byte address (low 32 bits of the generic shared pointer == LDS offset).
// ---------------------------------------------------------------------------
__device__ __forceinline__ void async_copy_b128(void* lds_dst,
                                                const void* uniform_base,
                                                unsigned byte_off)
{
    unsigned lds_addr = (unsigned)(unsigned long long)lds_dst;
    asm volatile("global_load_async_to_lds_b128 %0, %1, %2 offset:0"
                 :
                 : "v"(lds_addr), "v"(byte_off),
                   "s"((unsigned long long)uniform_base)
                 : "memory");
}
__device__ __forceinline__ void wait_asynccnt0()
{
    asm volatile("s_wait_asynccnt 0x0" ::: "memory");
}

// ---------------------------------------------------------------------------
// Prep 1: X (fp32 row-major) -> Xb (bf16 row-major). One-time, L2-resident.
// ---------------------------------------------------------------------------
__global__ __launch_bounds__(256)
void cvt_x_bf16(const float* __restrict__ X, __bf16* __restrict__ Xb)
{
    const int idx = (blockIdx.x * 256 + threadIdx.x) * 4;   // n = 1572864
    const float4 v = *reinterpret_cast<const float4*>(&X[idx]);
    v4bf o;
    o[0] = (__bf16)v.x; o[1] = (__bf16)v.y;
    o[2] = (__bf16)v.z; o[3] = (__bf16)v.w;
    *reinterpret_cast<v4bf*>(&Xb[idx]) = o;
}

// ---------------------------------------------------------------------------
// Prep 2: W (fp32, K x N row-major) -> WT (bf16, N x K "transposed") for all
// three projection matrices.  Makes B fragments contiguous-K in memory.
//   grid = (768, 1, 3); block row k = blockIdx.x of matrix blockIdx.z
// ---------------------------------------------------------------------------
__global__ __launch_bounds__(256)
void cvt_wT_bf16(const float* __restrict__ Wq, const float* __restrict__ Wk,
                 const float* __restrict__ Wv, __bf16* __restrict__ WT)
{
    const float* W = (blockIdx.z == 0) ? Wq : (blockIdx.z == 1) ? Wk : Wv;
    __bf16* WTz = WT + (size_t)blockIdx.z * WELEM;
    const int k = blockIdx.x;
    #pragma unroll
    for (int i = 0; i < 3; ++i) {
        const int col = threadIdx.x + i * 256;
        WTz[(size_t)col * HH + k] = (__bf16)W[(size_t)k * HH + col];
    }
}

// ---------------------------------------------------------------------------
// Kernel 1: QKV projection  Y = X @ W + b  via v_wmma_f32_16x16x32_bf16.
//   grid  = (HH/64, MROWS/128, 3); block = 256 threads = 8 waves.
//   Block tile 128(M) x 64(N); wave tile 16 x 64; K step 32.
//   Tiles staged bf16 via global_load_async_to_lds_b128; fragments are pure
//   ds_load_b128 pairs in the ISA 7.12.2 register layout -> no VALU cvt.
// ---------------------------------------------------------------------------
#define TM 128
#define TN 64
#define TK 32

__global__ __launch_bounds__(256, 2)
void qkv_wmma_gemm(const __bf16* __restrict__ Xb, const __bf16* __restrict__ WT,
                   const float* __restrict__ bq, const float* __restrict__ bk,
                   const float* __restrict__ bv,
                   float* __restrict__ Qo, float* __restrict__ Ko,
                   float* __restrict__ Vo)
{
    const float* bias; float* Y;
    if      (blockIdx.z == 0) { bias = bq; Y = Qo; }
    else if (blockIdx.z == 1) { bias = bk; Y = Ko; }
    else                      { bias = bv; Y = Vo; }
    const __bf16* WTz = WT + (size_t)blockIdx.z * WELEM;

    __shared__ __bf16 Xs[TM][TK];     // 128 x 32 bf16 = 8 KB (64B rows)
    __shared__ __bf16 Wt[TN][TK];     //  64 x 32 bf16 = 4 KB (rows = N cols)

    const int tid  = threadIdx.x;
    const int wave = tid >> 5;            // 0..7  -> M sub-tile
    const int lane = tid & 31;
    const int lm   = lane & 15;           // row (A) / col (B) in fragment
    const int hi   = lane >> 4;           // lane half selects K sub-range
    const int tileM = blockIdx.y * TM;
    const int tileN = blockIdx.x * TN;

    v8f acc[4] = {};

    // Per-K-step staging plan (16B chunks): X tile = 512 chunks (2/thread),
    // W tile = 256 chunks (1/thread).
    const int xr0 = tid >> 2;             // 0..63   (chunk -> row, seg)
    const int xs0 = tid & 3;
    const int wr  = tid >> 2;             // 0..63
    const int wsg = tid & 3;

    for (int kk = 0; kk < HH; kk += TK) {
        __syncthreads();                  // protect LDS reuse
        // X tile rows [tileM, tileM+128), K slice [kk, kk+32)
        async_copy_b128(&Xs[xr0][xs0 * 8], Xb,
                        (unsigned)(((tileM + xr0) * HH + kk + xs0 * 8) * 2));
        async_copy_b128(&Xs[xr0 + 64][xs0 * 8], Xb,
                        (unsigned)(((tileM + xr0 + 64) * HH + kk + xs0 * 8) * 2));
        // W^T tile rows (=output cols) [tileN, tileN+64), K slice [kk, kk+32)
        async_copy_b128(&Wt[wr][wsg * 8], WTz,
                        (unsigned)(((tileN + wr) * HH + kk + wsg * 8) * 2));
        wait_asynccnt0();
        __syncthreads();

        // A fragment (16-bit A 16x32): lanes 0-15 K={0..7,16..23},
        // lanes 16-31 K={8..15,24..31}; two contiguous 16B LDS loads.
        const v8bf* xp = reinterpret_cast<const v8bf*>(&Xs[wave * 16 + lm][0]);
        const v8bf alo = xp[hi];          // K = hi*8 .. +7
        const v8bf ahi = xp[hi + 2];      // K = hi*8+16 .. +23
        const v16bf afrag = __builtin_shufflevector(
            alo, ahi, 0, 1, 2, 3, 4, 5, 6, 7, 8, 9, 10, 11, 12, 13, 14, 15);

        #pragma unroll
        for (int n = 0; n < 4; ++n) {
            // B fragment: lanes 0-15 K={0..7,16..23}, lanes 16-31 K={8..15,24..31} for its col
            const v8bf* wp =
                reinterpret_cast<const v8bf*>(&Wt[n * 16 + lm][0]);
            const v8bf blo = wp[hi];
            const v8bf bhi = wp[hi + 2];
            const v16bf bfrag = __builtin_shufflevector(
                blo, bhi, 0, 1, 2, 3, 4, 5, 6, 7, 8, 9, 10, 11, 12, 13, 14, 15);
            acc[n] = __builtin_amdgcn_wmma_f32_16x16x32_bf16(
                         false, afrag, false, bfrag,
                         (short)0, acc[n], false, false);
        }
    }

    // Epilogue: bias add + store (C/D: VGPR r -> M = r + 8*laneHalf)
    #pragma unroll
    for (int n = 0; n < 4; ++n) {
        const int col = tileN + n * 16 + lm;
        const float bb = bias[col];
        #pragma unroll
        for (int rr = 0; rr < 8; ++rr) {
            const int row = tileM + wave * 16 + rr + hi * 8;
            Y[(size_t)row * HH + col] = acc[n][rr] + bb;
        }
    }
}

// ---------------------------------------------------------------------------
// Kernel 2: edge attention. One wave32 per node; 8 nodes per 256-thread block.
//   logits[e,h] = Q_h . (K[b,j_e] + Ek[r_e])_h / 8 ; wave32 shuffle softmax;
//   out = sum_e attn[e,h] * (V[b,j_e] + Ev[r_e])  (lane=column, coalesced).
// ---------------------------------------------------------------------------
__global__ __launch_bounds__(256, 4)
void edge_attn(const float* __restrict__ Qg, const float* __restrict__ Kg,
               const float* __restrict__ Vg, const int* __restrict__ eidx,
               const float* __restrict__ Ek, const float* __restrict__ Ev,
               float* __restrict__ out)
{
    const int wave = threadIdx.x >> 5;
    const int lane = threadIdx.x & 31;
    const int node = blockIdx.x * 8 + wave;     // 0..2047
    const int b    = node / NN;
    const int eb   = node * DEG;                // contiguous edge segment

    __shared__ float Qs[8][HH];                 // 24 KB
    __shared__ float attnS[8][DEG * NHD];       // 12 KB
    __shared__ int   jS[8][DEG];
    __shared__ int   rS[8][DEG];

    for (int t = lane; t < HH; t += 32)
        Qs[wave][t] = Qg[(size_t)node * HH + t];

    const int j = eidx[2 * EE + eb + lane];
    const int r = eidx[3 * EE + eb + lane];
    jS[wave][lane] = j;
    rS[wave][lane] = r;
    __syncthreads();

    // phase A: per-edge logits with fused K_node + E_k
    const float4* Q4 = reinterpret_cast<const float4*>(&Qs[wave][0]);
    const float4* K4 = reinterpret_cast<const float4*>(Kg + (size_t)(b * NN + j) * HH);
    const float4* E4 = reinterpret_cast<const float4*>(Ek + (size_t)r * HH);

    float lg[NHD];
    #pragma unroll
    for (int h = 0; h < NHD; ++h) {
        float acc = 0.f;
        #pragma unroll
        for (int q = 0; q < DH / 4; ++q) {
            const float4 qq = Q4[h * (DH / 4) + q];   // LDS broadcast
            const float4 kk = K4[h * (DH / 4) + q];
            const float4 ee = E4[h * (DH / 4) + q];
            acc = fmaf(qq.x, kk.x + ee.x, acc);
            acc = fmaf(qq.y, kk.y + ee.y, acc);
            acc = fmaf(qq.z, kk.z + ee.z, acc);
            acc = fmaf(qq.w, kk.w + ee.w, acc);
        }
        lg[h] = acc * 0.125f;                     // 1/sqrt(64)
    }

    // wave32 softmax per head across the 32 edges
    #pragma unroll
    for (int h = 0; h < NHD; ++h) {
        float m = lg[h];
        #pragma unroll
        for (int off = 16; off > 0; off >>= 1)
            m = fmaxf(m, __shfl_xor(m, off, 32));
        const float ex = __expf(lg[h] - m);
        float z = ex;
        #pragma unroll
        for (int off = 16; off > 0; off >>= 1)
            z += __shfl_xor(z, off, 32);
        attnS[wave][lane * NHD + h] = ex / z;
    }
    __syncthreads();

    // phase B: lane=column, coalesced V/Ev gathers
    const size_t outBase = (size_t)node * HH;
    #pragma unroll 2
    for (int c = 0; c < HH / 32; ++c) {           // 24 chunks of 32 columns
        const int d = c * 32 + lane;
        const int h = c >> 1;
        float acc = 0.f;
        #pragma unroll 8
        for (int e = 0; e < DEG; ++e) {
            const int   jj = jS[wave][e];
            const int   rr = rS[wave][e];
            const float a  = attnS[wave][e * NHD + h];
            acc = fmaf(a,
                       Vg[(size_t)(b * NN + jj) * HH + d] + Ev[(size_t)rr * HH + d],
                       acc);
        }
        out[outBase + d] = acc;
    }
}

// ---------------------------------------------------------------------------
// Launch.  Inputs (setup_inputs order):
//  0 node_states f32[2048*768]   1 edge_indices i32[4*65536]
//  2 Wq  3 bq  4 Wk  5 bk  6 Wv  7 bv   8 Ek f32[64*768]  9 Ev  10 num_heads
// Workspace: Q|K|V fp32 (18.9 MB) | Xbf16 (3.1 MB) | WT bf16 x3 (3.5 MB)
// ---------------------------------------------------------------------------
extern "C" void kernel_launch(void* const* d_in, const int* in_sizes, int n_in,
                              void* d_out, int out_size, void* d_ws, size_t ws_size,
                              hipStream_t stream)
{
    const float* X   = (const float*)d_in[0];
    const int*   ei  = (const int*)  d_in[1];
    const float* Wq  = (const float*)d_in[2];
    const float* bq  = (const float*)d_in[3];
    const float* Wk  = (const float*)d_in[4];
    const float* bk  = (const float*)d_in[5];
    const float* Wv  = (const float*)d_in[6];
    const float* bv  = (const float*)d_in[7];
    const float* Ek  = (const float*)d_in[8];
    const float* Ev  = (const float*)d_in[9];
    float*       out = (float*)d_out;

    const size_t perMat = (size_t)MROWS * HH;     // 1,572,864
    float*  Qw = (float*)d_ws;
    float*  Kw = Qw + perMat;
    float*  Vw = Kw + perMat;
    __bf16* Xb = (__bf16*)(Vw + perMat);
    __bf16* WT = Xb + perMat;                     // 3 x WELEM bf16

    // one-time bf16 conversion / transpose (L2-resident)
    cvt_x_bf16<<<dim3(perMat / 1024), dim3(256), 0, stream>>>(X, Xb);
    cvt_wT_bf16<<<dim3(HH, 1, 3), dim3(256), 0, stream>>>(Wq, Wk, Wv, WT);

    // 3 projection GEMMs through the WMMA pipe with async LDS staging
    qkv_wmma_gemm<<<dim3(HH / TN, MROWS / TM, 3), dim3(256), 0, stream>>>(
        Xb, WT, bq, bk, bv, Qw, Kw, Vw);

    // edge attention: 2048 nodes, one wave32 each
    edge_attn<<<dim3(MROWS / 8), dim3(256), 0, stream>>>(Qw, Kw, Vw, ei, Ek, Ev,
                                                         out);
}
